// SpikformerConv2dBNLIF_36756330119636
// MI455X (gfx1250) — matrix operand order repaired and checked by reading, not compiled, not run on hardware
//
#include <hip/hip_runtime.h>
#include <hip/hip_bf16.h>

// ---------------------------------------------------------------------------
// Spikformer Conv3x3+BN+MaxPool+LIF for MI455X (gfx1250).
//
//  prep       : x fp32 -> bf16 (once), W packed to per-K-chunk bf16 tiles
//  conv_bn    : implicit GEMM, M=Cout=128 x N=128 (two rows) per workgroup,
//               K=576 in 18 chunks of 32.  A-tile double-buffered and staged
//               by the Tensor Data Mover pipelined one chunk ahead
//               (tensor_load_to_lds overlapped with WMMA compute, closed by
//               s_wait_tensorcnt).  B-tile staged bf16 with halo zeros in a
//               16B-aligned layout so fragments load as ds_load_b128.
//               8x v_wmma_f32_16x16x32_bf16 per chunk per wave, BN fused.
//  maxpool_lif: MaxPool 3x3 s2 p1 fused with the 4-step LIF scan.
// ---------------------------------------------------------------------------

typedef __attribute__((ext_vector_type(16))) __bf16 v16bf;
typedef __attribute__((ext_vector_type(8)))  float  v8f;
typedef __attribute__((ext_vector_type(4)))  unsigned int u32x4;
typedef __attribute__((ext_vector_type(8)))  int          i32x8;
typedef __attribute__((ext_vector_type(4)))  int          i32x4;

#define TB    64      // T*B
#define CIN   64
#define COUT  128
#define HW    64      // H == W
#define KTOT  576     // CIN*9
#define OHW   32      // pooled spatial
#define NKC   18      // K chunks of 32
#define ATILE 4096    // 128*32 bf16 elements per packed A chunk
#define BPAD  40      // sB row stride in shorts: 80B -> 16B-aligned rows

#if defined(__has_builtin)
#  if __has_builtin(__builtin_amdgcn_tensor_load_to_lds)
#    define HAVE_TDM 1
#  endif
#endif

__device__ __forceinline__ unsigned f2bf(float f) {
  unsigned u = __float_as_uint(f);
  u += 0x7FFFu + ((u >> 16) & 1u);          // round-to-nearest-even
  return u >> 16;
}

#ifdef HAVE_TDM
// Issue a 1-D TDM copy of one packed A chunk (8KB contiguous) into LDS.
__device__ __forceinline__ void tdm_issue_a(const unsigned short* gsrc,
                                            void* lds_dst) {
  unsigned lds_addr = (unsigned)(__SIZE_TYPE__)lds_dst;
  unsigned long long ga = (unsigned long long)(__SIZE_TYPE__)gsrc;
  u32x4 g0 = { 1u,                                   // count=1, user D#
               lds_addr,                             // LDS byte address
               (unsigned)(ga & 0xFFFFFFFFu),         // global_addr lo
               (unsigned)((ga >> 32) & 0x1FFFFFFu) | (2u << 30) };  // type=2
  i32x8 g1 = { 0x00010000,        // data_size=1 (2 bytes/elem)
               0x20000000,        // tensor_dim0 lo16 (73728 & 0xFFFF)<<16
               0x00010001,        // tensor_dim0 hi16=1, tensor_dim1=1
               0x10000000,        // tile_dim0 = 4096
               0,                 // tile_dim1=0, tile_dim2=0 (1-D tile)
               ATILE,             // tensor_dim0_stride lo32
               0, 0 };
  i32x4 z4 = { 0, 0, 0, 0 };
#if defined(__clang_major__) && __clang_major__ >= 23
  i32x8 z8 = { 0, 0, 0, 0, 0, 0, 0, 0 };
  __builtin_amdgcn_tensor_load_to_lds(g0, g1, z4, z4, z8, 0);
#else
  __builtin_amdgcn_tensor_load_to_lds(g0, g1, z4, z4, 0);
#endif
}

__device__ __forceinline__ void tdm_wait0() {
#if defined(__has_builtin) && __has_builtin(__builtin_amdgcn_s_wait_tensorcnt)
  __builtin_amdgcn_s_wait_tensorcnt(0);
#else
  asm volatile("s_wait_tensorcnt 0x0" ::: "memory");
#endif
}
#endif  // HAVE_TDM

// ---- prep: x fp32->bf16 (4 elems/thread), W packed as [kc][m][kk] bf16 ----
__global__ __launch_bounds__(256)
void prep(const float* __restrict__ x, const float* __restrict__ w,
          unsigned short* __restrict__ xb, unsigned short* __restrict__ wb) {
  int idx = blockIdx.x * 256 + threadIdx.x;          // float4 index
  float4 v = reinterpret_cast<const float4*>(x)[idx];
  uint2 p;
  p.x = f2bf(v.x) | (f2bf(v.y) << 16);
  p.y = f2bf(v.z) | (f2bf(v.w) << 16);
  reinterpret_cast<uint2*>(xb)[idx] = p;

  if (idx < NKC * ATILE) {                           // 73728 weight elements
    int kc = idx / ATILE, r = idx & (ATILE - 1);
    int m = r >> 5, kk = r & 31;
    int tap = kc >> 1, ci0 = (kc & 1) << 5;
    wb[idx] = (unsigned short)f2bf(w[m * KTOT + (ci0 + kk) * 9 + tap]);
  }
}

// ---- conv + BN: one workgroup (8 waves) = M=128 x N=128 (rows oh0,oh0+1) ----
__global__ __launch_bounds__(256)
void conv_bn_wmma(const unsigned short* __restrict__ xb,
                  const unsigned short* __restrict__ wb,
                  const float* __restrict__ gamma, const float* __restrict__ beta,
                  const float* __restrict__ mean,  const float* __restrict__ var,
                  float* __restrict__ y) {
  __shared__ __align__(16) unsigned short sA[2][COUT * 32];  // double-buffered
  __shared__ __align__(16) unsigned short sB[128 * BPAD];    // [n][kk] padded
  __shared__ float sScale[COUT], sShift[COUT];

  const int tid  = threadIdx.x;
  const int wave = tid >> 5;
  const int lane = tid & 31;
  const int g    = lane >> 4;                // half-lane group (ISA layout)
  const int ln   = lane & 15;

  const int tb  = blockIdx.x >> 5;           // image index in [0,64)
  const int oh0 = (blockIdx.x & 31) << 1;    // first of two output rows

  if (tid < COUT) {                          // fold BN into scale/shift
    float s = gamma[tid] * rsqrtf(var[tid] + 1e-5f);
    sScale[tid] = s;
    sShift[tid] = beta[tid] - mean[tid] * s;
  }

  v8f acc[8] = {};                           // 8 n-tiles of 16x16 f32

#ifdef HAVE_TDM
  if (wave == 0) tdm_issue_a(wb, (void*)sA[0]);    // pipeline prologue
#endif

  for (int kc = 0; kc < NKC; ++kc) {
    const int buf = kc & 1;
    const int tap = kc >> 1;                 // conv tap 0..8
    const int ci0 = (kc & 1) << 5;           // cin half: 0 or 32
    const int kh  = tap / 3, kw = tap % 3;

    // ---- stage B: [n=128][kk=32] bf16 with halo zeros (16 elems/thread) ----
    #pragma unroll
    for (int i = 0; i < 16; ++i) {
      int idx = tid + i * 256;                 // 0..4095
      int n = idx & 127, kk = idx >> 7;        // n contiguous -> coalesced
      int ih = oh0 + (n >> 6) + kh - 1;
      int iw = (n & 63) + kw - 1;
      unsigned short v = 0;
      if ((unsigned)ih < (unsigned)HW && (unsigned)iw < (unsigned)HW)
        v = xb[((tb * CIN + ci0 + kk) * HW + ih) * HW + iw];
      sB[n * BPAD + kk] = v;
    }

#ifndef HAVE_TDM
    // fallback: manual copy of the packed A chunk
    #pragma unroll
    for (int i = 0; i < 16; ++i) {
      int idx = tid + i * 256;
      sA[buf][idx] = wb[kc * ATILE + idx];
    }
#endif

    // prefetch next chunk's activation rows (global_prefetch_b8)
    if (kc + 1 < NKC) {
      int ntap = (kc + 1) >> 1, nci0 = ((kc + 1) & 1) << 5;
      int nkh = ntap / 3, nkw = ntap % 3;
      int nn = tid & 127, nkk = tid >> 7;
      int nih = oh0 + (nn >> 6) + nkh - 1;
      int niw = (nn & 63) + nkw - 1;
      if ((unsigned)nih < (unsigned)HW && (unsigned)niw < (unsigned)HW)
        __builtin_prefetch(&xb[((tb * CIN + nci0 + nkk) * HW + nih) * HW + niw],
                           0, 1);
    }

#ifdef HAVE_TDM
    if (wave == 0) tdm_wait0();              // TDM(kc) complete
#endif
    __syncthreads();                         // sA[buf] + sB published

#ifdef HAVE_TDM
    // overlap: DMA next A chunk into the other buffer while we compute.
    // Its previous readers finished before the barrier above (compute kc-1).
    if (kc + 1 < NKC && wave == 0)
      tdm_issue_a(wb + (kc + 1) * ATILE, (void*)sA[buf ^ 1]);
#endif

    // ---- A fragment: 16x32 bf16, ISA 7.12.2 layout (2x ds_load_b128) ----
    // lane holds row m=ln; VGPR p holds K pair k0=(p>=4?16:0)+8g+2(p&3)
    union { v16bf v; u32x4 q[2]; } af;
    {
      const u32x4* arow = (const u32x4*)(sA[buf] + (wave * 16 + ln) * 32);
      af.q[0] = arow[g];                     // dwords g*4 .. g*4+3
      af.q[1] = arow[2 + g];                 // dwords 8+g*4 ..
    }

    // ---- 8 B fragments + WMMA ----
    #pragma unroll
    for (int nt = 0; nt < 8; ++nt) {
      // B 32x16: lane holds column n=ln; K order matches A (8g.., 16+8g..)
      union { v16bf v; u32x4 q[2]; } bfv;
      const u32x4* brow = (const u32x4*)(sB + (nt * 16 + ln) * BPAD);
      bfv.q[0] = brow[g];                    // dwords g*4 .. g*4+3
      bfv.q[1] = brow[2 + g];                // dwords 8+g*4 ..
      acc[nt] = __builtin_amdgcn_wmma_f32_16x16x32_bf16(
          false, af.v, false, bfv.v, (short)0, acc[nt], false, false);
    }
    __syncthreads();                         // done reading sB
  }

  // ---- epilogue: BN scale/shift, store pre-pool activations ----
  // C/D layout: lane -> (n = ln), VGPR r -> row m = 8g + r
  #pragma unroll
  for (int nt = 0; nt < 8; ++nt) {
    int n  = nt * 16 + ln;
    int oh = oh0 + (n >> 6);
    int ow = n & 63;
    #pragma unroll
    for (int r = 0; r < 8; ++r) {
      int co = wave * 16 + g * 8 + r;
      y[((tb * COUT + co) * HW + oh) * HW + ow] =
          acc[nt][r] * sScale[co] + sShift[co];
    }
  }
}

// ---- MaxPool 3x3 s2 p1 fused with 4-step LIF (tau=2, v_th=1, hard reset) ----
__global__ __launch_bounds__(256)
void maxpool_lif(const float* __restrict__ y, float* __restrict__ out) {
  int idx = blockIdx.x * 256 + threadIdx.x;  // [0, 16*128*32*32)
  int b   = idx >> 17;
  int rem = idx & 131071;
  int co  = rem >> 10;
  int pix = rem & 1023;
  int oh2 = pix >> 5, ow2 = pix & 31;

  float v = 0.f;
  #pragma unroll
  for (int t = 0; t < 4; ++t) {
    const float* base = y + (size_t)(((t * 16 + b) * COUT + co) * HW) * HW;
    float m = -3.402823466e38f;
    #pragma unroll
    for (int dy = 0; dy < 3; ++dy) {
      int ih = 2 * oh2 - 1 + dy;
      if ((unsigned)ih >= (unsigned)HW) continue;
      #pragma unroll
      for (int dx = 0; dx < 3; ++dx) {
        int iw = 2 * ow2 - 1 + dx;
        if ((unsigned)iw >= (unsigned)HW) continue;
        m = fmaxf(m, base[ih * HW + iw]);
      }
    }
    v = 0.5f * (v + m);                       // charge: v + (x - v)/tau
    float spike = (v >= 1.0f) ? 1.0f : 0.0f;  // fire
    out[(((t * 16 + b) * COUT + co) * OHW + oh2) * OHW + ow2] = spike;
    v = (spike > 0.f) ? 0.f : v;              // hard reset
  }
}

extern "C" void kernel_launch(void* const* d_in, const int* in_sizes, int n_in,
                              void* d_out, int out_size, void* d_ws, size_t ws_size,
                              hipStream_t stream) {
  (void)in_sizes; (void)n_in; (void)out_size; (void)ws_size;
  const float* x     = (const float*)d_in[0];  // [4,16,64,64,64]
  const float* w     = (const float*)d_in[1];  // [128,64,3,3]
  const float* gamma = (const float*)d_in[2];
  const float* beta  = (const float*)d_in[3];
  const float* mean  = (const float*)d_in[4];
  const float* var   = (const float*)d_in[5];

  // workspace layout:
  //   [0, 128MB)           y   : conv+BN output fp32 [64,128,64,64]
  //   [128MB, 160MB)       xb  : x in bf16
  //   [160MB, +144KB)      wb  : packed weights bf16 [18][128][32]
  float*          y  = (float*)d_ws;
  unsigned short* xb = (unsigned short*)((char*)d_ws + 134217728);
  unsigned short* wb = (unsigned short*)((char*)d_ws + 134217728 + 33554432);

  // 16.78M x elements / 4 per thread = 4,194,304 threads
  prep<<<dim3(16384), dim3(256), 0, stream>>>(x, w, xb, wb);

  // 2048 workgroups: one per (image, output-row pair)
  conv_bn_wmma<<<dim3(TB * (HW / 2)), dim3(256), 0, stream>>>(
      xb, wb, gamma, beta, mean, var, y);

  // 16*128*32*32 = 2,097,152 threads
  maxpool_lif<<<dim3((16 * COUT * OHW * OHW) / 256), dim3(256), 0, stream>>>(
      y, (float*)d_out);
}
